// GRUSingleRollout_48593259987103
// MI455X (gfx1250) — matrix operand order repaired and checked by reading, not compiled, not run on hardware
//
#include <hip/hip_runtime.h>

// ---------------------------------------------------------------------------
// GRU trajectory encoder + fused MLP head for MI455X (gfx1250, wave32, WMMA)
// ---------------------------------------------------------------------------

typedef __attribute__((ext_vector_type(16))) _Float16 v16h;
typedef __attribute__((ext_vector_type(8)))  _Float16 v8h;
typedef __attribute__((ext_vector_type(8)))  float    v8f;

#define T_STEPS   64
#define HD        64      // hidden per direction
#define G3        192     // 3*H gate columns
#define SEQ_TILE  16      // sequences (WMMA M) per workgroup
#define GRU_THREADS 384   // 12 wave32 -> one 16-col gate tile per wave

struct GruParams {
    const float* Wih[2][2];   // [layer][dir] (192 x D)
    const float* Whh[2][2];   // (192 x 64)
    const float* bih[2][2];   // (192)
    const float* bhh[2][2];   // (192)
};

// Dynamic LDS layout (bytes)
#define OFF_OF   0                         // layer0 fwd outputs: 64*16*64 f16
#define OFF_OB   131072                    // layer0 bwd outputs
#define OFF_SUM  262144                    // r/z pre-activations: 16*128 f32
#define OFF_GIN  270336                    // n-gate gi: 16*64 f32
#define OFF_GHN  274432                    // n-gate gh: 16*64 f32
#define OFF_H32  278528                    // hidden f32: 16*64
#define OFF_H16  282624                    // hidden f16: 16*64
#define OFF_X    284672                    // input xy: 16*64*2 f32
#define GRU_LDS_BYTES 292864               // ~286 KB < 320 KB/WGP

__device__ __forceinline__ float sigmoidf_(float x) {
    return 1.0f / (1.0f + __expf(-x));
}

// Async byte-copy global -> LDS, 16B per lane, tracked by ASYNCcnt (gfx1250).
// LDS byte address = low 32 bits of the generic pointer (ISA 10.2 aperture map).
__device__ __forceinline__ void async_copy_b128(void* lds_ptr,
                                                const void* gbase,
                                                unsigned byte_off) {
    unsigned ldsb = (unsigned)(size_t)lds_ptr;
    asm volatile("global_load_async_to_lds_b128 %0, %1, %2"
                 :
                 : "v"(ldsb), "v"(byte_off), "s"(gbase)
                 : "memory");
}
__device__ __forceinline__ void async_wait0() {
    asm volatile("s_wait_asynccnt 0x0" ::: "memory");
}

// Build a v16h A-fragment (16x32 f16) from LDS: contiguous layout,
// lane half h covers K = [16h, 16h+15] -> two 16B ds loads.
__device__ __forceinline__ v16h load_a_frag(const _Float16* rowptr, int half) {
    const _Float16* ap = rowptr + 16 * half;
    v8h lo = *(const v8h*)ap;
    v8h hi = *(const v8h*)(ap + 8);
    v16h a;
#pragma unroll
    for (int j = 0; j < 8; ++j) { a[j] = lo[j]; a[j + 8] = hi[j]; }
    return a;
}

__global__ __launch_bounds__(GRU_THREADS, 1)
void gru_encoder_kernel(const float* __restrict__ xg,             // (Nseq,64,2)
                        const unsigned char* __restrict__ maskg,  // (Nseq,64) bool
                        GruParams P,
                        float* __restrict__ hout)                 // (Nseq,128) [fwd|bwd]
{
    extern __shared__ char smem[];
    _Float16* of_lds = (_Float16*)(smem + OFF_OF);
    _Float16* ob_lds = (_Float16*)(smem + OFF_OB);
    float*    sum_s  = (float*)(smem + OFF_SUM);
    float*    gin_s  = (float*)(smem + OFF_GIN);
    float*    ghn_s  = (float*)(smem + OFF_GHN);
    float*    h32    = (float*)(smem + OFF_H32);
    _Float16* h16    = (_Float16*)(smem + OFF_H16);
    float*    x_lds  = (float*)(smem + OFF_X);

    const int tid  = threadIdx.x;
    const int wave = tid >> 5;
    const int lane = tid & 31;
    const int half = lane >> 4;
    const int lr   = lane & 15;
    const int seq0 = blockIdx.x * SEQ_TILE;
    const int ncol = wave * 16 + lr;          // owned gate column (0..191)

    // Stage the 16 sequences' xy history once via the async-DMA path
    // (8 KB = 512 x 16B chunks; no VGPR round-trip, tracked by ASYNCcnt).
    {
        const char* gsrc = (const char*)(xg + (size_t)seq0 * T_STEPS * 2);
        for (int c = tid; c < (SEQ_TILE * T_STEPS * 2 * 4) / 16; c += GRU_THREADS)
            async_copy_b128((char*)x_lds + c * 16, gsrc, (unsigned)(c * 16));
        async_wait0();
    }
    __syncthreads();

    for (int pass = 0; pass < 4; ++pass) {
        const int  l    = pass >> 1;
        const int  d    = pass & 1;
        const bool lay1 = (l == 1);
        const bool rev  = (d == 1);
        const float* Wih = P.Wih[l][d];
        const float* Whh = P.Whh[l][d];
        const float* bih = P.bih[l][d];
        const float* bhh = P.bhh[l][d];
        _Float16* o_lds = (d == 0) ? of_lds : ob_lds;

        __builtin_prefetch(Whh + (size_t)ncol * HD, 0, 3);
        if (lay1) __builtin_prefetch(Wih + (size_t)ncol * 128, 0, 3);

        // Recurrent weights Whh^T as resident B fragments (K=64 -> 2 chunks).
        // B layout: lane half h holds K = j + 16h within a 32-K chunk, col = lr.
        v16h bh[2];
#pragma unroll
        for (int c = 0; c < 2; ++c)
#pragma unroll
            for (int j = 0; j < 16; ++j)
                bh[c][j] = (_Float16)Whh[(size_t)ncol * HD + (j + 16 * half + 32 * c)];

        v16h bi[4];
        if (lay1) {
#pragma unroll
            for (int c = 0; c < 4; ++c)
#pragma unroll
                for (int j = 0; j < 16; ++j)
                    bi[c][j] = (_Float16)Wih[(size_t)ncol * 128 + (j + 16 * half + 32 * c)];
        }
        const float bih_c = bih[ncol];
        const float bhh_c = bhh[ncol];

        // h0 = 0
        for (int idx = tid; idx < SEQ_TILE * HD; idx += GRU_THREADS) {
            h32[idx] = 0.0f;
            h16[idx] = (_Float16)0.0f;
        }
        __syncthreads();

        for (int t = 0; t < T_STEPS; ++t) {
            const int tt = rev ? (T_STEPS - 1 - t) : t;

            if (!lay1) {
                // Layer 0 input dim = 2: gi is two FMAs per gate column.
                for (int idx = tid; idx < SEQ_TILE * G3; idx += GRU_THREADS) {
                    int m = idx / G3;
                    int col = idx - m * G3;
                    float x0 = x_lds[m * 128 + tt * 2 + 0];
                    float x1 = x_lds[m * 128 + tt * 2 + 1];
                    float gi = fmaf(x0, Wih[col * 2 + 0],
                               fmaf(x1, Wih[col * 2 + 1], bih[col]));
                    if (col < 128) sum_s[m * 128 + col] = gi;           // r,z
                    else           gin_s[m * HD + (col - 128)] = gi;    // n (gi part)
                }
                __syncthreads();
            }

            // gh = h @ Whh^T : 16x64x192, 2 WMMAs per wave
            v8f acc_h = {};
#pragma unroll
            for (int c = 0; c < 2; ++c) {
                v16h a = load_a_frag(&h16[lr * HD + 32 * c], half);
                acc_h = __builtin_amdgcn_wmma_f32_16x16x32_f16(
                    false, a, false, bh[c], (short)0, acc_h, false, false);
            }

            if (lay1) {
                // gi = inp @ Wih^T : inp = [of | ob] at time tt, K=128 -> 4 WMMAs
                v8f acc_i = {};
#pragma unroll
                for (int c = 0; c < 4; ++c) {
                    const _Float16* src = (c < 2) ? of_lds : ob_lds;
                    v16h a = load_a_frag(
                        &src[(tt * SEQ_TILE + lr) * HD + 32 * (c & 1)], half);
                    acc_i = __builtin_amdgcn_wmma_f32_16x16x32_f16(
                        false, a, false, bi[c], (short)0, acc_i, false, false);
                }
                if (wave < 8) {
#pragma unroll
                    for (int v = 0; v < 8; ++v) {
                        int m = v + 8 * half;
                        sum_s[m * 128 + ncol] = acc_i[v] + acc_h[v] + bih_c + bhh_c;
                    }
                } else {
#pragma unroll
                    for (int v = 0; v < 8; ++v) {
                        int m = v + 8 * half;
                        gin_s[m * HD + (ncol - 128)] = acc_i[v] + bih_c;
                        ghn_s[m * HD + (ncol - 128)] = acc_h[v] + bhh_c;
                    }
                }
            } else {
                if (wave < 8) {
#pragma unroll
                    for (int v = 0; v < 8; ++v) {
                        int m = v + 8 * half;
                        sum_s[m * 128 + ncol] += acc_h[v] + bhh_c;
                    }
                } else {
#pragma unroll
                    for (int v = 0; v < 8; ++v) {
                        int m = v + 8 * half;
                        ghn_s[m * HD + (ncol - 128)] = acc_h[v] + bhh_c;
                    }
                }
            }
            __syncthreads();

            // GRU cell update (mask-aware), 16x64 elements cooperatively
            for (int idx = tid; idx < SEQ_TILE * HD; idx += GRU_THREADS) {
                int m = idx >> 6;
                int c = idx & 63;
                float r  = sigmoidf_(sum_s[m * 128 + c]);
                float z  = sigmoidf_(sum_s[m * 128 + 64 + c]);
                float n  = tanhf(gin_s[idx] + r * ghn_s[idx]);
                float hp = h32[idx];
                float hn = (1.0f - z) * n + z * hp;
                bool  mt = maskg[(size_t)(seq0 + m) * T_STEPS + tt] != 0;
                float hv = mt ? hn : hp;
                h32[idx] = hv;
                _Float16 hh = (_Float16)hv;
                h16[idx] = hh;
                if (!lay1) o_lds[(tt * SEQ_TILE + m) * HD + c] = hh;
            }
            __syncthreads();
        }

        if (lay1) {
            for (int idx = tid; idx < SEQ_TILE * HD; idx += GRU_THREADS) {
                int m = idx >> 6;
                int c = idx & 63;
                hout[(size_t)(seq0 + m) * 128 + d * HD + c] = h32[idx];
            }
            __syncthreads();
        }
    }
}

// ---------------------------------------------------------------------------
// Fuse MLP + head: z=[h_ag|h_ego] (256) -> ReLU 128 -> ReLU 128 -> 60
// 16 rows per workgroup, 8 waves; each wave owns a 16-col output tile.
// ---------------------------------------------------------------------------
#define FUSE_THREADS 256

__global__ __launch_bounds__(FUSE_THREADS, 2)
void fuse_head_kernel(const float* __restrict__ h_ag,   // (8192,128)
                      const float* __restrict__ h_ego,  // (128,128)
                      const float* __restrict__ W1, const float* __restrict__ b1,
                      const float* __restrict__ W2, const float* __restrict__ b2,
                      const float* __restrict__ Wh, const float* __restrict__ bhd,
                      float* __restrict__ out)          // (8192,60)
{
    __shared__ _Float16 z0[16 * 256];
    __shared__ _Float16 z1[16 * 128];
    __shared__ _Float16 z2[16 * 128];

    const int tid  = threadIdx.x;
    const int wave = tid >> 5;
    const int lane = tid & 31;
    const int half = lane >> 4;
    const int lr   = lane & 15;
    const int r0   = blockIdx.x * 16;
    const int ncol = wave * 16 + lr;

    __builtin_prefetch(W1 + (size_t)ncol * 256, 0, 3);

    for (int idx = tid; idx < 16 * 256; idx += FUSE_THREADS) {
        int m = idx >> 8, k = idx & 255;
        int row = r0 + m;
        float v = (k < 128) ? h_ag[(size_t)row * 128 + k]
                            : h_ego[(size_t)(row >> 6) * 128 + (k - 128)];
        z0[idx] = (_Float16)v;
    }
    __syncthreads();

    // Stage 1: 256 -> 128, ReLU
    {
        v8f acc = {};
#pragma unroll
        for (int c = 0; c < 8; ++c) {
            v16h a = load_a_frag(&z0[lr * 256 + 32 * c], half);
            v16h b;
#pragma unroll
            for (int j = 0; j < 16; ++j)
                b[j] = (_Float16)W1[(size_t)ncol * 256 + (j + 16 * half + 32 * c)];
            acc = __builtin_amdgcn_wmma_f32_16x16x32_f16(
                false, a, false, b, (short)0, acc, false, false);
        }
        float bias = b1[ncol];
#pragma unroll
        for (int v = 0; v < 8; ++v) {
            int m = v + 8 * half;
            float y = acc[v] + bias;
            z1[m * 128 + ncol] = (_Float16)(y > 0.0f ? y : 0.0f);
        }
    }
    __syncthreads();

    // Stage 2: 128 -> 128, ReLU
    {
        v8f acc = {};
#pragma unroll
        for (int c = 0; c < 4; ++c) {
            v16h a = load_a_frag(&z1[lr * 128 + 32 * c], half);
            v16h b;
#pragma unroll
            for (int j = 0; j < 16; ++j)
                b[j] = (_Float16)W2[(size_t)ncol * 128 + (j + 16 * half + 32 * c)];
            acc = __builtin_amdgcn_wmma_f32_16x16x32_f16(
                false, a, false, b, (short)0, acc, false, false);
        }
        float bias = b2[ncol];
#pragma unroll
        for (int v = 0; v < 8; ++v) {
            int m = v + 8 * half;
            float y = acc[v] + bias;
            z2[m * 128 + ncol] = (_Float16)(y > 0.0f ? y : 0.0f);
        }
    }
    __syncthreads();

    // Head: 128 -> 60 (waves 0..3 cover 64 padded cols, mask to 60)
    if (wave < 4) {
        bool valid = ncol < 60;
        v8f acc = {};
#pragma unroll
        for (int c = 0; c < 4; ++c) {
            v16h a = load_a_frag(&z2[lr * 128 + 32 * c], half);
            v16h b;
#pragma unroll
            for (int j = 0; j < 16; ++j)
                b[j] = valid ? (_Float16)Wh[(size_t)ncol * 128 + (j + 16 * half + 32 * c)]
                             : (_Float16)0.0f;
            acc = __builtin_amdgcn_wmma_f32_16x16x32_f16(
                false, a, false, b, (short)0, acc, false, false);
        }
        if (valid) {
            float bias = bhd[ncol];
#pragma unroll
            for (int v = 0; v < 8; ++v) {
                int m = v + 8 * half;
                out[(size_t)(r0 + m) * 60 + ncol] = acc[v] + bias;
            }
        }
    }
}

// ---------------------------------------------------------------------------
extern "C" void kernel_launch(void* const* d_in, const int* in_sizes, int n_in,
                              void* d_out, int out_size, void* d_ws, size_t ws_size,
                              hipStream_t stream)
{
    const float*         agents_xy   = (const float*)d_in[0];
    const unsigned char* agents_mask = (const unsigned char*)d_in[1];
    const float*         ego_xy      = (const float*)d_in[2];
    const unsigned char* ego_mask    = (const unsigned char*)d_in[3];

    GruParams agP, egP;
    int i = 4;
    for (int l = 0; l < 2; ++l)
        for (int d = 0; d < 2; ++d) {
            agP.Wih[l][d] = (const float*)d_in[i++];
            agP.Whh[l][d] = (const float*)d_in[i++];
            agP.bih[l][d] = (const float*)d_in[i++];
            agP.bhh[l][d] = (const float*)d_in[i++];
        }
    for (int l = 0; l < 2; ++l)
        for (int d = 0; d < 2; ++d) {
            egP.Wih[l][d] = (const float*)d_in[i++];
            egP.Whh[l][d] = (const float*)d_in[i++];
            egP.bih[l][d] = (const float*)d_in[i++];
            egP.bhh[l][d] = (const float*)d_in[i++];
        }
    const float* W1  = (const float*)d_in[36];
    const float* b1  = (const float*)d_in[37];
    const float* W2  = (const float*)d_in[38];
    const float* b2  = (const float*)d_in[39];
    const float* Wh  = (const float*)d_in[40];
    const float* bhd = (const float*)d_in[41];

    float* h_ag  = (float*)d_ws;                   // 8192*128 f32 = 4 MB
    float* h_ego = h_ag + (size_t)8192 * 128;      // 128*128 f32

    // Agent encoder: 8192 sequences -> 512 workgroups of 16
    gru_encoder_kernel<<<512, GRU_THREADS, GRU_LDS_BYTES, stream>>>(
        agents_xy, agents_mask, agP, h_ag);
    // Ego encoder: 128 sequences -> 8 workgroups
    gru_encoder_kernel<<<8, GRU_THREADS, GRU_LDS_BYTES, stream>>>(
        ego_xy, ego_mask, egP, h_ego);
    // Fuse + head: 8192 rows -> 512 workgroups of 16
    fuse_head_kernel<<<512, FUSE_THREADS, 0, stream>>>(
        h_ag, h_ego, W1, b1, W2, b2, Wh, bhd, (float*)d_out);
}
